// Controller_65764539236401
// MI455X (gfx1250) — matrix operand order repaired and checked by reading, not compiled, not run on hardware
//
#include <hip/hip_runtime.h>
#include <hip/hip_bf16.h>
#include <math.h>

// ---------------------------------------------------------------------------
// ENAS LSTM controller sampler for gfx1250 (MI455X), wave32 + WMMA f16.
// One wave owns 16 batch rows for the full 40-step sequential program.
// Weights live in LDS pre-swizzled into WMMA B-operand layout; h/x live in a
// per-wave LDS f16 buffer in A-operand-friendly row-major layout; cell state c
// stays in registers in WMMA C-layout. Transcendentals use native gfx1250
// v_tanh_f32 / v_rcp_f32 / v_exp_f32 / v_log_f32.
// ---------------------------------------------------------------------------

#define NW      2          // waves per block
#define TEMP_   1.1f
#define TC_     2.5f

typedef __attribute__((ext_vector_type(16))) _Float16 v16h;
typedef __attribute__((ext_vector_type(8)))  _Float16 v8h;
typedef __attribute__((ext_vector_type(8)))  float    v8f;

#if __has_builtin(__builtin_amdgcn_tanhf)
__device__ __forceinline__ float tanh_(float x){ return __builtin_amdgcn_tanhf(x); }
#else
__device__ __forceinline__ float tanh_(float x){ return tanhf(x); }
#endif
#if __has_builtin(__builtin_amdgcn_rcpf)
__device__ __forceinline__ float rcp_(float x){ return __builtin_amdgcn_rcpf(x); }
#else
__device__ __forceinline__ float rcp_(float x){ return 1.0f/x; }
#endif
__device__ __forceinline__ float sigmoid_(float x){ return rcp_(1.0f+__expf(-x)); }

// -------- LDS layout (units: halves for f16 regions, floats for f32) --------
#define WL_H   32768   // [16 N-tiles][4 K-chunks][32 lanes][16 halves] fused W_ih;W_hh (K=128,N=256)
#define WP_H   4096    // w_prev  (K=64,N=64): [4][2][32][16]
#define WC_H   4096    // w_curr
#define WO_H   1024    // w_ops padded to N=16: [1][2][32][16]
#define EF_H   64      // embed_first f16
#define EO_H   320     // embed_ops f16 [5][64]
#define SH_HALVES (WL_H+WP_H+WC_H+WO_H+EF_H+EO_H)       // 42368
#define SH_FLOATS 468  // bgate256, bprev64, bcurr64, wout64, bops16, bout4
#define XH_H   2048    // per wave: x||h f16 [16 rows][128]
#define PLST_H 7168    // per wave: prev_lstm f16 [7][16][64]
#define PFC_H  7168    // per wave: prev_fc f16, C-layout packed [7][4][32][8]
#define SCR_F  256     // per wave: f32 scratch [16][16]
#define WSTR_H (XH_H+PLST_H+PFC_H+2*SCR_F)              // 16896 halves
#define SMEM_BYTES ((size_t)SH_HALVES*2 + (size_t)SH_FLOATS*4 + (size_t)NW*WSTR_H*2)

__device__ __forceinline__ v8f wmma_f16(v16h a, v16h b, v8f c){
  // emits v_wmma_f32_16x16x32_f16
  return __builtin_amdgcn_wmma_f32_16x16x32_f16(false, a, false, b, (short)0, c, false, false);
}

// A operand (16x32 f16) from row-major LDS buffer [16 rows][stride halves].
// ISA layout: lanes 0-15 row M=lane, K={0..7,16..23}; lanes 16-31 K={8..15,24..31}.
__device__ __forceinline__ v16h load_A(const _Float16* base, int lane, int kc, int stride){
  int m = lane & 15;
  int koff = kc*32 + ((lane & 16) ? 8 : 0);
  const _Float16* p = base + m*stride + koff;
  v8h lo = *(const v8h*)p;
  v8h hi = *(const v8h*)(p + 16);
  v16h r;
#pragma unroll
  for (int j=0;j<8;++j){ r[j]=lo[j]; r[j+8]=hi[j]; }
  return r;
}

// B operand: pre-swizzled in LDS, contiguous 16 halves per lane.
__device__ __forceinline__ v16h load_B(const _Float16* packed, int lane){
  return *(const v16h*)(packed + lane*16);
}

// one LSTM cell step: reads x||h f16 [16][128] from xh, updates c (C-layout)
// and writes new h (f16) into the h-region of xh (cols 64..127).
__device__ __forceinline__ void lstm_step(_Float16* xh, const _Float16* Wl,
                                          const float* bgate, v8f c[4], int lane){
  v16h A[4];
#pragma unroll
  for (int kc=0;kc<4;++kc) A[kc]=load_A(xh, lane, kc, 128);
  int col16 = lane & 15;
  int hi8   = (lane & 16) ? 8 : 0;
  _Float16* hout = xh + 64;
#pragma unroll
  for (int t=0;t<4;++t){
    float bi=bgate[      t*16+col16], bf=bgate[ 64+t*16+col16];
    float bg=bgate[128+t*16+col16],   bo=bgate[192+t*16+col16];
    v8f gi,gf,gg,go;
#pragma unroll
    for (int v=0;v<8;++v){ gi[v]=bi; gf[v]=bf; gg[v]=bg; go[v]=bo; }
#pragma unroll
    for (int kc=0;kc<4;++kc){
      gi = wmma_f16(A[kc], load_B(Wl + (( t    )*4+kc)*512, lane), gi);
      gf = wmma_f16(A[kc], load_B(Wl + ((t+ 4 )*4+kc)*512, lane), gf);
      gg = wmma_f16(A[kc], load_B(Wl + ((t+ 8 )*4+kc)*512, lane), gg);
      go = wmma_f16(A[kc], load_B(Wl + ((t+12 )*4+kc)*512, lane), go);
    }
#pragma unroll
    for (int v=0;v<8;++v){
      float cn = sigmoid_(gf[v])*c[t][v] + sigmoid_(gi[v])*tanh_(gg[v]);
      c[t][v] = cn;
      float hn = sigmoid_(go[v])*tanh_(cn);
      hout[(v+hi8)*128 + t*16 + col16] = (_Float16)hn;
    }
  }
  __builtin_amdgcn_wave_barrier();
}

// 64->64 projection of h (hbuf = xh+64, stride 128): out = h @ W.T + b, C-layout.
__device__ __forceinline__ void proj64(const _Float16* hbuf, const _Float16* W,
                                       const float* bias, v8f out[4], int lane){
  v16h A0=load_A(hbuf,lane,0,128), A1=load_A(hbuf,lane,1,128);
  int col16 = lane & 15;
#pragma unroll
  for (int t=0;t<4;++t){
    float b = bias[t*16+col16];
    v8f acc;
#pragma unroll
    for (int v=0;v<8;++v) acc[v]=b;
    acc = wmma_f16(A0, load_B(W+(t*2+0)*512,lane), acc);
    acc = wmma_f16(A1, load_B(W+(t*2+1)*512,lane), acc);
    out[t]=acc;
  }
}

__device__ __forceinline__ void store_pfc(_Float16* pfc, int entry, const v8f out[4], int lane){
#pragma unroll
  for (int t=0;t<4;++t){
    v8h pk;
#pragma unroll
    for (int v=0;v<8;++v) pk[v]=(_Float16)out[t][v];
    *(v8h*)(pfc + ((entry*4+t)*32+lane)*8) = pk;
  }
  __builtin_amdgcn_wave_barrier();
}

// broadcast a 64-half vector (same for every row) into x-region of xh
__device__ __forceinline__ void set_input_row(_Float16* xh, int lane, const _Float16* src64){
  int r = lane&15, off = (lane&16)?32:0;
  _Float16* d = xh + r*128 + off;
  const _Float16* s = src64 + off;
  *(v8h*)(d)    = *(const v8h*)(s);
  *(v8h*)(d+8)  = *(const v8h*)(s+8);
  *(v8h*)(d+16) = *(const v8h*)(s+16);
  *(v8h*)(d+24) = *(const v8h*)(s+24);
  __builtin_amdgcn_wave_barrier();
}

// per-row selected 64-half vector into x-region (table[a][(r)][64] or table[a][64])
__device__ __forceinline__ void set_input_sel(_Float16* xh, int lane, const _Float16* table,
                                              int entry_stride, int a, bool per_row){
  int r = lane&15, off = (lane&16)?32:0;
  const _Float16* s = table + a*entry_stride + (per_row ? r*64 : 0) + off;
  _Float16* d = xh + r*128 + off;
  *(v8h*)(d)    = *(const v8h*)(s);
  *(v8h*)(d+8)  = *(const v8h*)(s+8);
  *(v8h*)(d+16) = *(const v8h*)(s+16);
  *(v8h*)(d+24) = *(const v8h*)(s+24);
  __builtin_amdgcn_wave_barrier();
}

// save h (xh cols 64..127) into prev_lstm[n]
__device__ __forceinline__ void save_h(_Float16* plst, const _Float16* xh, int n, int lane){
  int r = lane&15, off = (lane&16)?32:0;
  const _Float16* s = xh + r*128 + 64 + off;
  _Float16* d = plst + n*1024 + r*64 + off;
  *(v8h*)(d)    = *(const v8h*)(s);
  *(v8h*)(d+8)  = *(const v8h*)(s+8);
  *(v8h*)(d+16) = *(const v8h*)(s+16);
  *(v8h*)(d+24) = *(const v8h*)(s+24);
  __builtin_amdgcn_wave_barrier();
}

// scalar softmax / gumbel-argmax over n<=6 logits held in lg[] (lane<16 only).
// Fully unrolled with k<n guards so lg[] stays in registers (no scratch).
__device__ __forceinline__ int sample_from_logits(float* lg, int n, const float* noise,
    int step, int bsz, int myrow, int lane, float& lp, float& ent, float* out,
    const float* extra /*nullptr or ops_extra*/){
  int a = 0;
  if (lane < 16){
    float mx = -1e30f;
#pragma unroll
    for (int k=0;k<6;++k) if (k<n){
      float l = TC_ * tanh_(lg[k] * (1.0f/TEMP_));
      if (extra) l += extra[k];
      lg[k] = l;
      mx = fmaxf(mx, l);
    }
    float se = 0.f;
#pragma unroll
    for (int k=0;k<6;++k) if (k<n) se += __expf(lg[k]-mx);
    float lse = mx + __logf(se);
    float best = -1e30f, blp = 0.f;
#pragma unroll
    for (int k=0;k<6;++k) if (k<n){
      float logp = lg[k]-lse;
      ent += -logp * __expf(logp);
      float u = noise[(size_t)step*(size_t)bsz*7 + (size_t)myrow*7 + k];
      u = fminf(fmaxf(u, 1e-7f), 1.0f-1e-7f);
      float per = logp - __logf(-__logf(u));   // logp + gumbel
      if (per > best){ best = per; a = k; blp = logp; }
    }
    lp += blp;
    out[(size_t)myrow*40 + step] = (float)a;
  }
  return __shfl(a, lane & 15);
}

__global__ __launch_bounds__(NW*32) void enas_controller_kernel(
    const float* __restrict__ noise,       const float* __restrict__ embed_first,
    const float* __restrict__ embed_ops,   const float* __restrict__ w_ih,
    const float* __restrict__ w_hh,        const float* __restrict__ b_ih,
    const float* __restrict__ b_hh,        const float* __restrict__ w_prev,
    const float* __restrict__ b_prev,      const float* __restrict__ w_curr,
    const float* __restrict__ b_curr,      const float* __restrict__ w_out,
    const float* __restrict__ b_out,       const float* __restrict__ w_ops,
    const float* __restrict__ b_ops,       float* __restrict__ out, int bsz)
{
  extern __shared__ float4 smem_f4[];
  _Float16* Wl = (_Float16*)smem_f4;          // fused [W_ih;W_hh] B-operand layout
  _Float16* Wp = Wl + WL_H;
  _Float16* Wc = Wp + WP_H;
  _Float16* Wo = Wc + WC_H;
  _Float16* ef = Wo + WO_H;
  _Float16* eo = ef + EF_H;
  float* bgate = (float*)(eo + EO_H);
  float* bprev = bgate + 256;
  float* bcurr = bprev + 64;
  float* woutS = bcurr + 64;
  float* bopsS = woutS + 64;
  float* boutS = bopsS + 16;
  _Float16* wavebase = (_Float16*)(boutS + 4);

  const int tid = threadIdx.x, nthr = blockDim.x;

  // ---- pack weights into WMMA B-operand layout (f32 -> f16) ----
  for (int i=tid; i<WL_H; i+=nthr){
    int j=i&15, lane=(i>>4)&31, kc=(i>>9)&3, t=i>>11;
    int K = kc*32 + ((lane&16)?8:0) + (j<8 ? j : j+8);       // input dim (x:0..63, h:64..127)
    int N = t*16 + (lane&15);            // gate column 0..255 (i,f,g,o)
    float v = (K<64) ? w_ih[N*64+K] : w_hh[N*64+(K-64)];
    Wl[i] = (_Float16)v;
  }
  for (int i=tid; i<WP_H; i+=nthr){
    int j=i&15, lane=(i>>4)&31, kc=(i>>9)&1, t=i>>10;
    int K = kc*32 + ((lane&16)?8:0) + (j<8 ? j : j+8);
    int N = t*16 + (lane&15);
    Wp[i] = (_Float16)w_prev[N*64+K];
    Wc[i] = (_Float16)w_curr[N*64+K];
  }
  for (int i=tid; i<WO_H; i+=nthr){
    int j=i&15, lane=(i>>4)&31, kc=(i>>9)&1;
    int K = kc*32 + ((lane&16)?8:0) + (j<8 ? j : j+8);
    int N = lane&15;
    Wo[i] = (N<5) ? (_Float16)w_ops[N*64+K] : (_Float16)0.0f;
  }
  for (int i=tid;i<64;i+=nthr){
    ef[i]=(_Float16)embed_first[i];
    bprev[i]=b_prev[i]; bcurr[i]=b_curr[i]; woutS[i]=w_out[i];
  }
  for (int i=tid;i<EO_H;i+=nthr) eo[i]=(_Float16)embed_ops[i];
  for (int i=tid;i<256;i+=nthr)  bgate[i]=b_ih[i]+b_hh[i];
  for (int i=tid;i<16;i+=nthr)   bopsS[i]=(i<5)? b_ops[i] : 0.0f;
  if (tid==0) boutS[0]=b_out[0];
  // zero per-wave xh (h region must start 0) and prev_lstm entries 0,1
  for (int i=tid; i<NW*XH_H; i+=nthr){
    int w=i/XH_H, off=i%XH_H;
    wavebase[w*WSTR_H + off] = (_Float16)0.0f;
  }
  for (int i=tid; i<NW*2048; i+=nthr){
    int w=i/2048, off=i%2048;
    wavebase[w*WSTR_H + XH_H + off] = (_Float16)0.0f;   // prev_lstm[0..1]
  }
  __syncthreads();

  // ---- per-wave pointers / row ownership ----
  const int wave = tid >> 5;
  const int lane = tid & 31;
  _Float16* xh   = wavebase + wave*WSTR_H;
  _Float16* plst = xh + XH_H;
  _Float16* pfc  = plst + PLST_H;
  float*    scr  = (float*)(pfc + PFC_H);
  const int rowbase = blockIdx.x*(NW*16) + wave*16;
  const int myrow   = rowbase + (lane & 15);
  const int col16   = lane & 15;
  const int hi8     = (lane & 16) ? 8 : 0;
  const float ops_extra[5] = {0.25f,0.25f,-0.25f,-0.25f,-0.25f};

  v8f c[4];
#pragma unroll
  for (int t=0;t<4;++t)
#pragma unroll
    for (int v=0;v<8;++v) c[t][v]=0.f;
  float lp=0.f, ent=0.f;
  int step=0;

  for (int cell=0; cell<2; ++cell){
    int n = 2;
    set_input_row(xh, lane, ef);                 // inputs = first_in
    for (int t2=0; t2<2; ++t2){                  // two warm-up LSTM calls
      lstm_step(xh, Wl, bgate, c, lane);
      v8f p4[4]; proj64(xh+64, Wp, bprev, p4, lane);
      store_pfc(pfc, t2, p4, lane);              // prev_lstm[t2] stays zero (torch bug)
    }
    for (int node=2; node<7; ++node){
      // ---- two index samples ----
      for (int j=0;j<2;++j){
        lstm_step(xh, Wl, bgate, c, lane);
        v8f curr[4]; proj64(xh+64, Wc, bcurr, curr, lane);
        float lg[6];
#pragma unroll
        for (int k=0;k<6;++k) if (k<n){
          float psum[8];
#pragma unroll
          for (int v=0;v<8;++v) psum[v]=0.f;
#pragma unroll
          for (int t=0;t<4;++t){
            v8h pf = *(const v8h*)(pfc + ((k*4+t)*32+lane)*8);
            float wv = woutS[t*16+col16];
#pragma unroll
            for (int v=0;v<8;++v){
              float q = tanh_((float)pf[v] + curr[t][v]);
              psum[v] += q*wv;
            }
          }
#pragma unroll
          for (int v=0;v<8;++v) scr[(v+hi8)*16 + col16] = psum[v];
          __builtin_amdgcn_wave_barrier();
          asm volatile("" ::: "memory");
          if (lane<16){
            float s=0.f;
#pragma unroll
            for (int u=0;u<16;++u) s += scr[lane*16+u];
            lg[k] = s + boutS[0];
          }
          asm volatile("" ::: "memory");
          __builtin_amdgcn_wave_barrier();
        }
        int aa = sample_from_logits(lg, n, noise, step, bsz, myrow, lane,
                                    lp, ent, out, nullptr);
        set_input_sel(xh, lane, plst, 1024, aa, true);   // inputs = prev_lstm[a]
        ++step;
      }
      // ---- two op samples ----
      for (int j=0;j<2;++j){
        lstm_step(xh, Wl, bgate, c, lane);
        v16h A0 = load_A(xh+64, lane, 0, 128);
        v16h A1 = load_A(xh+64, lane, 1, 128);
        float b = bopsS[col16];
        v8f acc;
#pragma unroll
        for (int v=0;v<8;++v) acc[v]=b;
        acc = wmma_f16(A0, load_B(Wo,      lane), acc);
        acc = wmma_f16(A1, load_B(Wo+512,  lane), acc);
#pragma unroll
        for (int v=0;v<8;++v) scr[(v+hi8)*16 + col16] = acc[v];
        __builtin_amdgcn_wave_barrier();
        asm volatile("" ::: "memory");
        float lg[6];
        if (lane<16){
#pragma unroll
          for (int k=0;k<5;++k) lg[k] = scr[lane*16+k];
        }
        asm volatile("" ::: "memory");
        __builtin_amdgcn_wave_barrier();
        int aa = sample_from_logits(lg, 5, noise, step, bsz, myrow, lane,
                                    lp, ent, out, ops_extra);
        set_input_sel(xh, lane, eo, 64, aa, false);      // inputs = embed_ops[a]
        ++step;
      }
      // ---- node epilogue ----
      lstm_step(xh, Wl, bgate, c, lane);
      save_h(plst, xh, n, lane);
      v8f p4[4]; proj64(xh+64, Wp, bprev, p4, lane);
      store_pfc(pfc, n, p4, lane);
      ++n;
      set_input_row(xh, lane, ef);                       // inputs = first_in
    }
  }

  if (lane < 16){
    out[(size_t)bsz*40 + myrow]               = lp;
    out[(size_t)bsz*40 + (size_t)bsz + myrow] = ent;
  }
}

extern "C" void kernel_launch(void* const* d_in, const int* in_sizes, int n_in,
                              void* d_out, int out_size, void* d_ws, size_t ws_size,
                              hipStream_t stream) {
  (void)d_ws; (void)ws_size; (void)n_in; (void)out_size;
  const float* noise       = (const float*)d_in[0];
  const float* embed_first = (const float*)d_in[1];
  const float* embed_ops   = (const float*)d_in[2];
  const float* w_ih        = (const float*)d_in[3];
  const float* w_hh        = (const float*)d_in[4];
  const float* b_ih        = (const float*)d_in[5];
  const float* b_hh        = (const float*)d_in[6];
  const float* w_prev      = (const float*)d_in[7];
  const float* b_prev      = (const float*)d_in[8];
  const float* w_curr      = (const float*)d_in[9];
  const float* b_curr      = (const float*)d_in[10];
  const float* w_out       = (const float*)d_in[11];
  const float* b_out       = (const float*)d_in[12];
  const float* w_ops       = (const float*)d_in[13];
  const float* b_ops       = (const float*)d_in[14];
  float* out = (float*)d_out;

  const int bsz = in_sizes[0] / (40*7);          // noise is [40, B, 7]
  const int rows_per_block = NW*16;
  dim3 grid((bsz + rows_per_block - 1) / rows_per_block);
  dim3 block(NW*32);
  enas_controller_kernel<<<grid, block, SMEM_BYTES, stream>>>(
      noise, embed_first, embed_ops, w_ih, w_hh, b_ih, b_hh,
      w_prev, b_prev, w_curr, b_curr, w_out, b_out, w_ops, b_ops,
      out, bsz);
}
